// GptModel_13529146983074
// MI455X (gfx1250) — compile-verified, matrix operand order checked by reading, not run on hardware
//
#include <hip/hip_runtime.h>

// ---------------------------------------------------------------------------
// GPT forward for MI455X (gfx1250, wave32). All matmuls via
// v_wmma_f32_16x16x32_bf16 (fp32 weights converted to bf16 once per launch,
// fp32 accumulate). Attention is flash-style (no TxT buffer).
// GEMM: 256 threads / 8 waves, 128x128 C tile, 32-wide K steps, LDS
// double-buffered with async global->LDS copies (ASYNCcnt) when available;
// each wave computes 32x64 (2 A-frags x 4 B-frags = 8 WMMA per K step).
// ---------------------------------------------------------------------------

#define Bc  8
#define Tc  1024
#define Dc  1024
#define Hc  16
#define HSc 64
#define Lc  8
#define FFc 4096
#define Vc  96
#define BTc (Bc * Tc)

typedef __bf16 bf16;
typedef __attribute__((ext_vector_type(16))) __bf16 v16bf;
typedef __attribute__((ext_vector_type(8)))  float  v8f;
typedef __attribute__((ext_vector_type(4)))  int    v4i;

union Frag   { uint4 u[2]; v16bf v; };
union Pack16 { uint4 u[2]; bf16 h[16]; };

#if defined(__gfx1250__) && \
    __has_builtin(__builtin_amdgcn_global_load_async_to_lds_b128) && \
    __has_builtin(__builtin_amdgcn_s_wait_asynccnt)
#define USE_ASYNC_LDS 1
#else
#define USE_ASYNC_LDS 0
#endif

// Copy 16 bytes global -> LDS for this lane (async when supported).
__device__ inline void copy16_g2l(const void* g, void* l) {
#if USE_ASYNC_LDS
  __builtin_amdgcn_global_load_async_to_lds_b128(
      (__attribute__((address_space(1))) v4i*)g,
      (__attribute__((address_space(3))) v4i*)l, 0, 0);
#else
  *(uint4*)l = *(const uint4*)g;
#endif
}

__device__ inline void wait_async_copies() {
#if USE_ASYNC_LDS
  __builtin_amdgcn_s_wait_asynccnt(0);
#endif
}

__device__ inline v8f zero8() {
  v8f z;
#pragma unroll
  for (int i = 0; i < 8; ++i) z[i] = 0.0f;
  return z;
}

__device__ inline v8f wmma_bf16(v16bf a, v16bf b, v8f c) {
  // D = A(16x32) * B(32x16) + C, fp32 accumulate
  return __builtin_amdgcn_wmma_f32_16x16x32_bf16(
      /*neg_a=*/false, a, /*neg_b=*/false, b,
      /*c_mod=*/(short)0, c, /*reuse_a=*/false, /*reuse_b=*/false);
}

// ---------------------------------------------------------------------------
// Weight preparation
// ---------------------------------------------------------------------------

// W* [L,H,D,HS] fp32 -> [L, D, ldDst] bf16 slab at column h*HS+e (dst already
// offset to this projection's column block; ldDst = 3*D for packed QKV).
__global__ void repack_hde_kernel(const float* __restrict__ src,
                                  bf16* __restrict__ dst, long ldDst, long n) {
  long stride = (long)gridDim.x * blockDim.x;
  for (long i = (long)blockIdx.x * blockDim.x + threadIdx.x; i < n; i += stride) {
    long e = i % HSc;
    long d = (i / HSc) % Dc;
    long h = (i / ((long)HSc * Dc)) % Hc;
    long l = i / ((long)HSc * Dc * Hc);
    dst[(l * Dc + d) * ldDst + h * HSc + e] = (bf16)src[i];
  }
}

__global__ void cvt_kernel(const float* __restrict__ src,
                           bf16* __restrict__ dst, long n) {
  long stride = (long)gridDim.x * blockDim.x;
  for (long i = (long)blockIdx.x * blockDim.x + threadIdx.x; i < n; i += stride)
    dst[i] = (bf16)src[i];
}

// ---------------------------------------------------------------------------
// Embedding:  x[b,t,:] = tok_emb[index[b,t]] + pos_emb[t]
// ---------------------------------------------------------------------------
__global__ __launch_bounds__(256) void embed_kernel(
    const int* __restrict__ idx, const float* __restrict__ tok,
    const float* __restrict__ pos, float* __restrict__ x,
    bf16* __restrict__ xb) {
  int row = blockIdx.x, tid = threadIdx.x;
  int t  = row % Tc;
  int tk = idx[row];
#pragma unroll
  for (int i = 0; i < 4; ++i) {
    int c = tid + i * 256;
    float v = tok[(size_t)tk * Dc + c] + pos[(size_t)t * Dc + c];
    x[(size_t)row * Dc + c]  = v;
    xb[(size_t)row * Dc + c] = (bf16)v;
  }
}

// ---------------------------------------------------------------------------
// Tiled WMMA GEMM: C[M,N] = A[M,K](bf16) * B[K,N](bf16) (+bias)(+resid)(ReLU?)
// 256 threads = 8 waves; 128x128 C tile; K-step 32; LDS double-buffered.
// M % 128 == 0, K % 32 == 0.
// Wave w: rows 32*(w>>1)..+31, cols 64*(w&1)..+63 -> 2x4 accumulators.
// ---------------------------------------------------------------------------
__global__ __launch_bounds__(256) void gemm_kernel(
    const bf16* __restrict__ A, const bf16* __restrict__ B,
    float* __restrict__ Cf, bf16* __restrict__ Cb,
    const float* __restrict__ bias, const float* __restrict__ resid,
    int M, int N, int K, int relu) {
  __shared__ __align__(16) bf16 As[2][128][32];   // A tiles, row-major
  __shared__ __align__(16) bf16 Bt[2][128][32];   // B tiles, transposed [n][k]

  const int t    = threadIdx.x;
  const int wave = t >> 5;
  const int lane = t & 31;
  const int l16  = lane & 15;
  const int lh   = lane >> 4;              // 0|1 -> K half-chunk select
  const int wr   = wave >> 1;              // wave row 0..3  (32 rows each)
  const int wc   = wave & 1;               // wave col 0..1  (64 cols each)
  const int rowBase = blockIdx.y * 128;
  const int colBase = blockIdx.x * 128;
  const bool fullN = (colBase + 128) <= N;

  v8f acc[2][4];
#pragma unroll
  for (int i = 0; i < 2; ++i)
#pragma unroll
    for (int s = 0; s < 4; ++s) acc[i][s] = zero8();

  const int arow = t >> 1;                 // 0..127
  const int ach  = (t & 1) * 16;           // 0|16
  const int bk   = t & 31;                 // 0..31
  const int bnb  = (t >> 5) * 16;          // 0,16,...,112

  auto loadTile = [&](int k0, int buf) {
    // A tile: 128 rows x 32 halves, 2x16B per thread, async when available
    const bf16* ap = A + (size_t)(rowBase + arow) * K + k0 + ach;
    copy16_g2l(ap,     &As[buf][arow][ach]);
    copy16_g2l(ap + 8, &As[buf][arow][ach + 8]);
    if (fullN) {  // B tile fast path: vector loads, branch-free scatter
      Pack16 w;
      const bf16* bp = B + (size_t)(k0 + bk) * N + colBase + bnb;
      w.u[0] = ((const uint4*)bp)[0];
      w.u[1] = ((const uint4*)bp)[1];
      __builtin_prefetch(bp + 32 * (size_t)N, 0, 3);  // next K tile of B
#pragma unroll
      for (int j = 0; j < 16; ++j) Bt[buf][bnb + j][bk] = w.h[j];
    } else {      // boundary tile: clamped address + select, no exec branches
      const bf16* brow = B + (size_t)(k0 + bk) * N;
#pragma unroll
      for (int j = 0; j < 16; ++j) {
        int gn = colBase + bnb + j;
        int gc = (gn < N) ? gn : (N - 1);
        bf16 v = brow[gc];
        if (gn >= N) v = (bf16)0.0f;
        Bt[buf][bnb + j][bk] = v;
      }
    }
  };

  const int nT = K >> 5;
  loadTile(0, 0);
  wait_async_copies();
  __syncthreads();

  for (int ti = 0; ti < nT; ++ti) {
    const int cur = ti & 1;
    if (ti + 1 < nT) loadTile((ti + 1) << 5, cur ^ 1);  // overlap with WMMA

    Frag af[2];
#pragma unroll
    for (int i = 0; i < 2; ++i) {
      const int r = wr * 32 + i * 16 + l16;
      af[i].u[0] = *((const uint4*)&As[cur][r][lh * 8]);
      af[i].u[1] = *((const uint4*)&As[cur][r][lh * 8 + 16]);
    }
#pragma unroll
    for (int s = 0; s < 4; ++s) {
      Frag bfr;
      const int n = wc * 64 + s * 16 + l16;
      bfr.u[0] = *((const uint4*)&Bt[cur][n][lh * 8]);
      bfr.u[1] = *((const uint4*)&Bt[cur][n][lh * 8 + 16]);
      acc[0][s] = wmma_bf16(af[0].v, bfr.v, acc[0][s]);
      acc[1][s] = wmma_bf16(af[1].v, bfr.v, acc[1][s]);
    }
    wait_async_copies();
    __syncthreads();
  }

#pragma unroll
  for (int i = 0; i < 2; ++i)
#pragma unroll
    for (int s = 0; s < 4; ++s)
#pragma unroll
      for (int ii = 0; ii < 8; ++ii) {
        int row = rowBase + wr * 32 + i * 16 + ii + 8 * lh;
        int col = colBase + wc * 64 + s * 16 + l16;
        if (col < N) {
          float v = acc[i][s][ii];
          if (bias)  v += bias[col];
          if (resid) v += resid[(size_t)row * N + col];
          if (relu)  v = fmaxf(v, 0.0f);
          if (Cf) Cf[(size_t)row * N + col] = v;
          if (Cb) Cb[(size_t)row * N + col] = (bf16)v;
        }
      }
}

// ---------------------------------------------------------------------------
// Flash attention: one wave per (b, h, 16-query tile). Q/K/V bf16 rows of
// length `ld` (packed qkv buffer), head slice at column h*HS. O bf16 [BT, D].
// ---------------------------------------------------------------------------
__global__ __launch_bounds__(32) void attn_kernel(
    const bf16* __restrict__ Q, const bf16* __restrict__ Km,
    const bf16* __restrict__ Vm, bf16* __restrict__ O, int ld) {
  const int lane = threadIdx.x;
  const int l16  = lane & 15;
  const int lh   = lane >> 4;
  const int qt = blockIdx.x, h = blockIdx.y, b = blockIdx.z;
  const int qBase = qt * 16;
  const size_t headOff = (size_t)h * HSc;

  __shared__ __align__(16) bf16 Pl[16][32];
  __shared__ __align__(16) bf16 Vl[32][64];

  Frag qf[2];
#pragma unroll
  for (int kk = 0; kk < 2; ++kk) {
    const bf16* qp = Q + (size_t)(b * Tc + qBase + l16) * ld + headOff +
                     kk * 32 + lh * 8;
    qf[kk].u[0] = *((const uint4*)qp);
    qf[kk].u[1] = *((const uint4*)(qp + 16));
  }

  v8f oacc[4];
#pragma unroll
  for (int u = 0; u < 4; ++u) oacc[u] = zero8();
  float m_run[8], l_run[8];
#pragma unroll
  for (int i = 0; i < 8; ++i) { m_run[i] = -1e30f; l_run[i] = 0.0f; }

  const int nChunks = (qBase + 16 + 31) >> 5;
  for (int ct = 0; ct < nChunks; ++ct) {
    const int kt = ct * 32;
    {  // stage V chunk [32 keys x 64] into LDS (async 16B copies per lane)
      const bf16* vp = Vm + (size_t)(b * Tc + kt + lane) * ld + headOff;
#pragma unroll
      for (int j = 0; j < 8; ++j)
        copy16_g2l(vp + j * 8, &Vl[lane][j * 8]);
    }
    // S = q * k^T  (16 x 32 keys), K = HS = 64 -> 2 wmma per 16-key subtile
    v8f sacc[2];
    sacc[0] = zero8(); sacc[1] = zero8();
#pragma unroll
    for (int s = 0; s < 2; ++s)
#pragma unroll
      for (int kk = 0; kk < 2; ++kk) {
        Frag kf;
        const bf16* kp = Km + (size_t)(b * Tc + kt + s * 16 + l16) * ld +
                         headOff + kk * 32 + lh * 8;
        kf.u[0] = *((const uint4*)kp);
        kf.u[1] = *((const uint4*)(kp + 16));
        sacc[s] = wmma_bf16(qf[kk].v, kf.v, sacc[s]);
      }
    // scale + causal mask (D layout: lane holds rows i+8*lh, col l16)
    float pv[2][8];
#pragma unroll
    for (int s = 0; s < 2; ++s)
#pragma unroll
      for (int i = 0; i < 8; ++i) {
        int key  = kt + s * 16 + l16;
        int qrow = qBase + i + 8 * lh;
        float sv = sacc[s][i] * 0.125f;  // HS^-0.5
        pv[s][i] = (key > qrow) ? -1e30f : sv;
      }
    // online softmax update (row reduction across 16-lane halves)
    float alpha[8];
#pragma unroll
    for (int i = 0; i < 8; ++i) {
      float cm = fmaxf(pv[0][i], pv[1][i]);
      cm = fmaxf(cm, __shfl_xor(cm, 1, 32));
      cm = fmaxf(cm, __shfl_xor(cm, 2, 32));
      cm = fmaxf(cm, __shfl_xor(cm, 4, 32));
      cm = fmaxf(cm, __shfl_xor(cm, 8, 32));
      float mn = fmaxf(m_run[i], cm);
      alpha[i] = __expf(m_run[i] - mn);
      m_run[i] = mn;
      float p0 = __expf(pv[0][i] - mn);
      float p1 = __expf(pv[1][i] - mn);
      pv[0][i] = p0; pv[1][i] = p1;
      float rs = p0 + p1;
      rs += __shfl_xor(rs, 1, 32);
      rs += __shfl_xor(rs, 2, 32);
      rs += __shfl_xor(rs, 4, 32);
      rs += __shfl_xor(rs, 8, 32);
      l_run[i] = l_run[i] * alpha[i] + rs;
    }
#pragma unroll
    for (int u = 0; u < 4; ++u)
#pragma unroll
      for (int i = 0; i < 8; ++i) oacc[u][i] *= alpha[i];

    wait_async_copies();   // V chunk resident before LDS reads
    __syncthreads();
    // P: D layout -> LDS -> A layout
#pragma unroll
    for (int s = 0; s < 2; ++s)
#pragma unroll
      for (int i = 0; i < 8; ++i)
        Pl[i + 8 * lh][s * 16 + l16] = (bf16)pv[s][i];
    __syncthreads();

    Frag pf;
    pf.u[0] = *((const uint4*)&Pl[l16][lh * 8]);
    pf.u[1] = *((const uint4*)&Pl[l16][lh * 8 + 16]);
    // O += P(16x32) * V(32x64): 4 N-subtiles; B frag = column gather from LDS
#pragma unroll
    for (int u = 0; u < 4; ++u) {
      v16bf vv;
      int n = u * 16 + l16;
#pragma unroll
      for (int j = 0; j < 8; ++j) {
        vv[j]     = Vl[lh * 8 + j][n];
        vv[8 + j] = Vl[lh * 8 + 16 + j][n];
      }
      oacc[u] = wmma_bf16(pf.v, vv, oacc[u]);
    }
    __syncthreads();
  }

#pragma unroll
  for (int u = 0; u < 4; ++u)
#pragma unroll
    for (int i = 0; i < 8; ++i) {
      int qrow = qBase + i + 8 * lh;
      float ov = oacc[u][i] / l_run[i];
      O[(size_t)(b * Tc + qrow) * Dc + headOff + u * 16 + l16] = (bf16)ov;
    }
}

// ---------------------------------------------------------------------------
// LayerNorm over D=1024 per row; writes fp32 (optional) and bf16 copies.
// ---------------------------------------------------------------------------
__global__ __launch_bounds__(256) void ln_kernel(
    const float* __restrict__ in, const float* __restrict__ g,
    const float* __restrict__ bb, float* __restrict__ outf,
    bf16* __restrict__ outb) {
  __shared__ float sm[256];
  int row = blockIdx.x, tid = threadIdx.x;
  const float* rp = in + (size_t)row * Dc;
  float v[4], s = 0.0f, s2 = 0.0f;
#pragma unroll
  for (int i = 0; i < 4; ++i) {
    v[i] = rp[tid + i * 256];
    s += v[i]; s2 += v[i] * v[i];
  }
  sm[tid] = s; __syncthreads();
  for (int off = 128; off > 0; off >>= 1) {
    if (tid < off) sm[tid] += sm[tid + off];
    __syncthreads();
  }
  float mean = sm[0] * (1.0f / Dc); __syncthreads();
  sm[tid] = s2; __syncthreads();
  for (int off = 128; off > 0; off >>= 1) {
    if (tid < off) sm[tid] += sm[tid + off];
    __syncthreads();
  }
  float var  = sm[0] * (1.0f / Dc) - mean * mean;
  float rstd = rsqrtf(var + 1e-5f);
#pragma unroll
  for (int i = 0; i < 4; ++i) {
    int c = tid + i * 256;
    float nv = (v[i] - mean) * rstd * g[c] + bb[c];
    if (outf) outf[(size_t)row * Dc + c] = nv;
    outb[(size_t)row * Dc + c] = (bf16)nv;
  }
}

// ---------------------------------------------------------------------------
// Loss: per-row log-softmax NLL over V=96, then deterministic mean reduce.
// ---------------------------------------------------------------------------
__global__ __launch_bounds__(128) void nll_kernel(
    const float* __restrict__ logits, const int* __restrict__ tgt,
    float* __restrict__ nll) {
  __shared__ float sm[128];
  int row = blockIdx.x, tid = threadIdx.x;
  const float* lp = logits + (size_t)row * Vc;
  float v = (tid < Vc) ? lp[tid] : -3.4e38f;
  sm[tid] = v; __syncthreads();
  for (int off = 64; off > 0; off >>= 1) {
    if (tid < off) sm[tid] = fmaxf(sm[tid], sm[tid + off]);
    __syncthreads();
  }
  float mx = sm[0]; __syncthreads();
  float e = (tid < Vc) ? __expf(v - mx) : 0.0f;
  sm[tid] = e; __syncthreads();
  for (int off = 64; off > 0; off >>= 1) {
    if (tid < off) sm[tid] += sm[tid + off];
    __syncthreads();
  }
  if (tid == 0) {
    float lse = __logf(sm[0]) + mx;
    nll[row] = lse - lp[tgt[row]];
  }
}

__global__ __launch_bounds__(256) void mean_kernel(
    const float* __restrict__ nll, int n, float* __restrict__ out) {
  __shared__ float sm[256];
  float s = 0.0f;
  for (int i = threadIdx.x; i < n; i += 256) s += nll[i];
  sm[threadIdx.x] = s; __syncthreads();
  for (int off = 128; off > 0; off >>= 1) {
    if (threadIdx.x < off) sm[threadIdx.x] += sm[threadIdx.x + off];
    __syncthreads();
  }
  if (threadIdx.x == 0) out[0] = sm[0] / (float)n;
}

// ---------------------------------------------------------------------------
// Host orchestration
// ---------------------------------------------------------------------------
extern "C" void kernel_launch(void* const* d_in, const int* in_sizes, int n_in,
                              void* d_out, int out_size, void* d_ws,
                              size_t ws_size, hipStream_t stream) {
  (void)in_sizes; (void)n_in; (void)out_size; (void)ws_size;

  const int*   index   = (const int*)d_in[0];
  const int*   targets = (const int*)d_in[1];
  const float* tok_emb = (const float*)d_in[2];
  const float* pos_emb = (const float*)d_in[3];
  const float* Wq  = (const float*)d_in[4];
  const float* Wk  = (const float*)d_in[5];
  const float* Wv  = (const float*)d_in[6];
  const float* Wo  = (const float*)d_in[7];
  const float* bo  = (const float*)d_in[8];
  const float* W1  = (const float*)d_in[9];
  const float* b1  = (const float*)d_in[10];
  const float* W2  = (const float*)d_in[11];
  const float* b2  = (const float*)d_in[12];
  const float* ln1_g = (const float*)d_in[13];
  const float* ln1_b = (const float*)d_in[14];
  const float* ln2_g = (const float*)d_in[15];
  const float* ln2_b = (const float*)d_in[16];
  const float* lnf_g = (const float*)d_in[17];
  const float* lnf_b = (const float*)d_in[18];
  const float* Wlm = (const float*)d_in[19];
  const float* blm = (const float*)d_in[20];
  float* out = (float*)d_out;

  // Workspace carve (256B aligned slabs)
  char* p = (char*)d_ws;
  auto take = [&p](size_t bytes) {
    char* r = p;
    p += (bytes + 255) & ~(size_t)255;
    return r;
  };
  const size_t DD  = (size_t)Dc * Dc;
  const size_t D3  = (size_t)Dc * 3;
  bf16* wqkvc = (bf16*)take((size_t)Lc * Dc * D3 * 2);  // [L][D][3D]
  bf16* woc   = (bf16*)take((size_t)Lc * DD * 2);
  bf16* w1c   = (bf16*)take((size_t)Lc * Dc * FFc * 2);
  bf16* w2c   = (bf16*)take((size_t)Lc * FFc * Dc * 2);
  bf16* wlmc  = (bf16*)take((size_t)Dc * Vc * 2);
  float* x    = (float*)take((size_t)BTc * Dc * 4);
  bf16*  xb   = (bf16*)take((size_t)BTc * Dc * 2);
  bf16*  qkvb = (bf16*)take((size_t)BTc * D3 * 2);      // [BT][3D]
  bf16*  ob   = (bf16*)take((size_t)BTc * Dc * 2);
  float* tbuf = (float*)take((size_t)BTc * Dc * 4);
  bf16*  h1   = (bf16*)take((size_t)BTc * FFc * 2);
  float* nllw = (float*)take((size_t)BTc * 4);

  // ---- weight conversion (once per launch; inputs never mutated) ----
  {
    long n = (long)Lc * Hc * Dc * HSc;
    int blocks = (int)((n + 255) / 256);
    repack_hde_kernel<<<blocks, 256, 0, stream>>>(Wq, wqkvc + 0 * Dc, (long)D3, n);
    repack_hde_kernel<<<blocks, 256, 0, stream>>>(Wk, wqkvc + 1 * Dc, (long)D3, n);
    repack_hde_kernel<<<blocks, 256, 0, stream>>>(Wv, wqkvc + 2 * Dc, (long)D3, n);
  }
  {
    long n = (long)Lc * DD;
    cvt_kernel<<<(int)((n + 255) / 256), 256, 0, stream>>>(Wo, woc, n);
  }
  {
    long n = (long)Lc * Dc * FFc;
    cvt_kernel<<<(int)((n + 255) / 256), 256, 0, stream>>>(W1, w1c, n);
    cvt_kernel<<<(int)((n + 255) / 256), 256, 0, stream>>>(W2, w2c, n);
  }
  {
    long n = (long)Dc * Vc;
    cvt_kernel<<<(int)((n + 255) / 256), 256, 0, stream>>>(Wlm, wlmc, n);
  }

  // ---- embedding ----
  embed_kernel<<<BTc, 256, 0, stream>>>(index, tok_emb, pos_emb, x, xb);

  const dim3 gemmBlk(256);
  const dim3 gridQKV((3 * Dc) / 128, BTc / 128);
  const dim3 gridD(Dc / 128, BTc / 128);
  const dim3 gridFF(FFc / 128, BTc / 128);
  const dim3 gridV((Vc + 127) / 128, BTc / 128);
  const dim3 attnGrid(Tc / 16, Hc, Bc);

  for (int l = 0; l < Lc; ++l) {
    // fused q|k|v projection -> qkvb [BT, 3D] bf16
    gemm_kernel<<<gridQKV, gemmBlk, 0, stream>>>(
        xb, wqkvc + (size_t)l * Dc * D3, nullptr, qkvb,
        nullptr, nullptr, BTc, 3 * Dc, Dc, 0);
    // attention -> ob (bf16, heads concatenated)
    attn_kernel<<<attnGrid, 32, 0, stream>>>(qkvb, qkvb + Dc, qkvb + 2 * Dc,
                                             ob, 3 * Dc);
    // t = ob @ Wo + bo + x  (fp32)
    gemm_kernel<<<gridD, gemmBlk, 0, stream>>>(
        ob, woc + (size_t)l * DD, tbuf, nullptr,
        bo + (size_t)l * Dc, x, BTc, Dc, Dc, 0);
    // x = LN(t)
    ln_kernel<<<BTc, 256, 0, stream>>>(tbuf, ln1_g + (size_t)l * Dc,
                                       ln1_b + (size_t)l * Dc, x, xb);
    // h1 = relu(x @ W1 + b1)  (bf16)
    gemm_kernel<<<gridFF, gemmBlk, 0, stream>>>(
        xb, w1c + (size_t)l * Dc * FFc, nullptr, h1,
        b1 + (size_t)l * FFc, nullptr, BTc, FFc, Dc, 1);
    // t = h1 @ W2 + b2 + x
    gemm_kernel<<<gridD, gemmBlk, 0, stream>>>(
        h1, w2c + (size_t)l * FFc * Dc, tbuf, nullptr,
        b2 + (size_t)l * Dc, x, BTc, Dc, FFc, 0);
    // x = LN(t)
    ln_kernel<<<BTc, 256, 0, stream>>>(tbuf, ln2_g + (size_t)l * Dc,
                                       ln2_b + (size_t)l * Dc, x, xb);
  }

  // final LN -> xb only
  ln_kernel<<<BTc, 256, 0, stream>>>(x, lnf_g, lnf_b, nullptr, xb);

  // logits = xb @ Wlm + blm  -> d_out (fp32, rows of 96)
  gemm_kernel<<<gridV, gemmBlk, 0, stream>>>(xb, wlmc, out, nullptr, blm,
                                             nullptr, BTc, Vc, Dc, 0);

  // loss
  nll_kernel<<<BTc, 128, 0, stream>>>(out, targets, nllw);
  mean_kernel<<<1, 256, 0, stream>>>(nllw, BTc, out + (size_t)BTc * Vc);
}
